// SelfAttention_67937792688325
// MI455X (gfx1250) — matrix operand order repaired and checked by reading, not compiled, not run on hardware
//
#include <hip/hip_runtime.h>
#include <hip/hip_bf16.h>
#include <stdint.h>

// Problem constants (match reference)
constexpr int B = 8, S = 2048, D = 512, P = 64;

typedef __attribute__((ext_vector_type(16))) __bf16 v16bf;
typedef __attribute__((ext_vector_type(8)))  float  v8f;

union V16BF {
    v16bf    v;
    uint32_t u[8];
};

// Pack two f32 -> packed bf16x2 dword via native casts (compiler picks lowering)
__device__ __forceinline__ uint32_t pack_bf16x2(float lo, float hi) {
    union { __bf16 h[2]; uint32_t u; } c;
    c.h[0] = (__bf16)lo;
    c.h[1] = (__bf16)hi;
    return c.u;
}

__device__ __forceinline__ uint16_t f32_to_bf16(float f) {
    union { __bf16 h; uint16_t u; } c;
    c.h = (__bf16)f;
    return c.u;
}

// ---------------------------------------------------------------------------
// Kernel 0: convert query fp32 -> bf16 [B][S][D] (pure bandwidth pass).
// ---------------------------------------------------------------------------
__global__ __launch_bounds__(256, 1)
void prep_query(const float* __restrict__ q, uint16_t* __restrict__ qbf) {
    int i = blockIdx.x * blockDim.x + threadIdx.x;   // over B*S*D/4 (exact)
    float4 f = ((const float4*)q)[i];
    uint2 o;
    o.x = pack_bf16x2(f.x, f.y);
    o.y = pack_bf16x2(f.z, f.w);
    ((uint2*)qbf)[i] = o;
}

// ---------------------------------------------------------------------------
// Kernel 1: transpose + convert weights to bf16 (B-operand-friendly layouts)
//   WqT/WkT/WvT : [P][D]  (B[k=d][n=p] reads are contiguous dwords)
//   WoT         : [D][P]  (B[k=p][n=d] reads are contiguous dwords)
// ---------------------------------------------------------------------------
__global__ void prep_weights(const float* __restrict__ Wq, const float* __restrict__ Wk,
                             const float* __restrict__ Wv, const float* __restrict__ Wo,
                             uint16_t* __restrict__ WqT, uint16_t* __restrict__ WkT,
                             uint16_t* __restrict__ WvT, uint16_t* __restrict__ WoT) {
    int i = blockIdx.x * blockDim.x + threadIdx.x;   // over D*P
    if (i >= D * P) return;
    int d = i / P, p = i % P;
    WqT[(size_t)p * D + d] = f32_to_bf16(Wq[(size_t)d * P + p]);
    WkT[(size_t)p * D + d] = f32_to_bf16(Wk[(size_t)d * P + p]);
    WvT[(size_t)p * D + d] = f32_to_bf16(Wv[(size_t)d * P + p]);
    WoT[(size_t)d * P + p] = f32_to_bf16(Wo[(size_t)p * D + d]);
}

// ---------------------------------------------------------------------------
// Kernel 2: QKV projection.  One wave per (which in {q,k,v}, 16-row s-block).
//   Reads pre-converted bf16 query; pure loads + WMMA in the hot loop.
//   q,k stored row-major bf16 [B][S][P];  v stored transposed bf16 [B][P][S].
// ---------------------------------------------------------------------------
__global__ __launch_bounds__(256, 1)
void qkv_kernel(const uint16_t* __restrict__ qbf,
                const uint16_t* __restrict__ WqT, const uint16_t* __restrict__ WkT,
                const uint16_t* __restrict__ WvT,
                const float* __restrict__ bq, const float* __restrict__ bk,
                const float* __restrict__ bv,
                uint16_t* __restrict__ qb, uint16_t* __restrict__ kb,
                uint16_t* __restrict__ vT) {
    const int wave = threadIdx.x >> 5;
    const int lane = threadIdx.x & 31;
    const int n = lane & 15, h = lane >> 4;

    int task  = blockIdx.x * 8 + wave;   // 384 blocks * 8 waves = 3072 = 3*B*(S/16)
    int which = task >> 10;              // 0=q 1=k 2=v   (1024 tasks each)
    int rem   = task & 1023;
    int b     = rem >> 7;                // S/16 = 128
    int sblk  = rem & 127;

    const uint16_t* WT = (which == 0) ? WqT : (which == 1) ? WkT : WvT;
    const float* bias  = (which == 0) ? bq  : (which == 1) ? bk  : bv;

    v8f c[4] = {};
    // A-operand row for this lane: M = n  ->  s row
    const uint16_t* qrowb = qbf + ((size_t)b * S + sblk * 16 + n) * D;

    for (int kc = 0; kc < 16; ++kc) {              // K = D = 512, chunks of 32
        V16BF a;
#pragma unroll
        for (int v = 0; v < 8; ++v) {
            int K = kc * 32 + h * 16 + 2 * v;   // A layout
            a.u[v] = *(const uint32_t*)(qrowb + K);
        }
        V16BF bm[4];
#pragma unroll
        for (int nt = 0; nt < 4; ++nt) {
            const uint16_t* wr = WT + (size_t)(nt * 16 + n) * D;     // col p = nt*16+n
#pragma unroll
            for (int v = 0; v < 8; ++v) {
                int K = kc * 32 + h * 16 + 2 * v;                    // B layout
                bm[nt].u[v] = *(const uint32_t*)(wr + K);
            }
        }
#pragma unroll
        for (int nt = 0; nt < 4; ++nt)
            c[nt] = __builtin_amdgcn_wmma_f32_16x16x32_bf16(
                false, a.v, false, bm[nt].v, (short)0, c[nt], false, false);
    }

    // Wave-uniform branch around whole store blocks
    if (which == 0) {
#pragma unroll
        for (int nt = 0; nt < 4; ++nt) {
            int p = nt * 16 + n;
            float bi = bias[p];
#pragma unroll
            for (int r = 0; r < 8; ++r) {
                int s = sblk * 16 + r + 8 * h;                    // C layout M=r+8h
                qb[((size_t)b * S + s) * P + p] = f32_to_bf16(c[nt][r] + bi);
            }
        }
    } else if (which == 1) {
#pragma unroll
        for (int nt = 0; nt < 4; ++nt) {
            int p = nt * 16 + n;
            float bi = bias[p];
#pragma unroll
            for (int r = 0; r < 8; ++r) {
                int s = sblk * 16 + r + 8 * h;
                kb[((size_t)b * S + s) * P + p] = f32_to_bf16(c[nt][r] + bi);
            }
        }
    } else {
        // vT[b][p][s]: consecutive rows r,r+1 adjacent -> packed dword stores
#pragma unroll
        for (int nt = 0; nt < 4; ++nt) {
            int p = nt * 16 + n;
            float bi = bias[p];
            uint16_t* dst = vT + ((size_t)b * P + p) * S + sblk * 16 + 8 * h;
#pragma unroll
            for (int r = 0; r < 8; r += 2) {
                *(uint32_t*)(dst + r) = pack_bf16x2(c[nt][r] + bi, c[nt][r + 1] + bi);
            }
        }
    }
}

// ---------------------------------------------------------------------------
// Kernel 3: flash attention + fused output projection.
//   One wave per (b, 16-row s-block). Streams over t in chunks of 32 with
//   online softmax; probs C->A layout conversion via per-wave 2KB LDS tile;
//   final x Wo + bo fused.
// ---------------------------------------------------------------------------
__global__ __launch_bounds__(128, 1)
void attn_kernel(const uint16_t* __restrict__ qb, const uint16_t* __restrict__ kb,
                 const uint16_t* __restrict__ vT, const uint16_t* __restrict__ WoT,
                 const float* __restrict__ bo, const float* __restrict__ mask,
                 float* __restrict__ out) {
    __shared__ uint16_t smem[4 * 1024];            // 2KB per wave
    const int wave = threadIdx.x >> 5;
    const int lane = threadIdx.x & 31;
    const int n = lane & 15, h = lane >> 4;

    int task = blockIdx.x * 4 + wave;              // 256 blocks * 4 = 1024 row-blocks
    int b    = task >> 7;
    int sblk = task & 127;
    uint16_t* myLds = smem + wave * 1024;

    // Q A-operand tiles (16x64 -> two 16x32 tiles), resident for whole kernel
    const uint16_t* qrow = qb + ((size_t)b * S + sblk * 16 + n) * P;   // M = n
    V16BF aq[2];
#pragma unroll
    for (int t = 0; t < 2; ++t)
#pragma unroll
        for (int v = 0; v < 8; ++v) {
            int K = t * 32 + h * 16 + 2 * v;
            aq[t].u[v] = *(const uint32_t*)(qrow + K);
        }

    // Row-wise mask term (reference broadcasts mask over last axis: row mask)
    float maskadd[8];
#pragma unroll
    for (int r = 0; r < 8; ++r) {
        float mv = mask[(size_t)b * S + sblk * 16 + r + 8 * h];
        maskadd[r] = (1.0f - mv) * -10000.0f;
    }

    float mrow[8], lrow[8];
    v8f acc[4] = {};
#pragma unroll
    for (int r = 0; r < 8; ++r) { mrow[r] = -1e30f; lrow[r] = 0.0f; }

    for (int tb = 0; tb < S; tb += 32) {
        // Prefetch next chunk of K and V (global_prefetch_b8)
        if (tb + 32 < S) {
            __builtin_prefetch(kb + ((size_t)b * S + tb + 32 + n) * P, 0, 1);
            __builtin_prefetch(vT + ((size_t)b * P + n) * S + tb + 32, 0, 1);
        }

        // ---- load all K B-tiles for this chunk, then the 4 score WMMAs ----
        V16BF bt[4];                   // [tt*2 + (lo/hi)]
#pragma unroll
        for (int tt = 0; tt < 2; ++tt) {
            const uint16_t* krow = kb + ((size_t)b * S + tb + tt * 16 + n) * P; // N=t col
#pragma unroll
            for (int v = 0; v < 8; ++v) {
                int K = h * 16 + 2 * v;                 // B layout, K = p
                bt[tt * 2 + 0].u[v] = *(const uint32_t*)(krow + K);
                bt[tt * 2 + 1].u[v] = *(const uint32_t*)(krow + 32 + K);
            }
        }
        v8f sc[2];
#pragma unroll
        for (int tt = 0; tt < 2; ++tt) {
            v8f z = {};
            z = __builtin_amdgcn_wmma_f32_16x16x32_bf16(false, aq[0].v, false,
                    bt[tt * 2 + 0].v, (short)0, z, false, false);
            z = __builtin_amdgcn_wmma_f32_16x16x32_bf16(false, aq[1].v, false,
                    bt[tt * 2 + 1].v, (short)0, z, false, false);
            sc[tt] = z;
        }

        // ---- issue V B-tile loads now; softmax VALU below hides their latency
#pragma unroll
        for (int nt = 0; nt < 4; ++nt) {
            const uint16_t* vrow = vT + ((size_t)b * P + nt * 16 + n) * S;  // N=p col
#pragma unroll
            for (int v = 0; v < 8; ++v) {
                int K = h * 16 + 2 * v;                 // K = t offset
                bt[nt].u[v] = *(const uint32_t*)(vrow + tb + K);
            }
        }

        // ---- online softmax update (per row = r + 8h) ----
        float p0[8], p1[8], alpha[8];
#pragma unroll
        for (int r = 0; r < 8; ++r) {
            float v0 = sc[0][r] * 0.125f + maskadd[r];  // 1/sqrt(P)=0.125
            float v1 = sc[1][r] * 0.125f + maskadd[r];
            float cm = fmaxf(v0, v1);
#pragma unroll
            for (int m = 1; m < 16; m <<= 1) cm = fmaxf(cm, __shfl_xor(cm, m, 32));
            float nm = fmaxf(mrow[r], cm);
            float a  = __expf(mrow[r] - nm);
            float e0 = __expf(v0 - nm), e1 = __expf(v1 - nm);
            float rs = e0 + e1;
#pragma unroll
            for (int m = 1; m < 16; m <<= 1) rs += __shfl_xor(rs, m, 32);
            lrow[r] = lrow[r] * a + rs;
            mrow[r] = nm;
            alpha[r] = a; p0[r] = e0; p1[r] = e1;
        }
#pragma unroll
        for (int nt = 0; nt < 4; ++nt)
#pragma unroll
            for (int r = 0; r < 8; ++r) acc[nt][r] *= alpha[r];

        // ---- probs C-layout -> LDS (row-major 16x32 bf16) -> A-layout ----
#pragma unroll
        for (int r = 0; r < 8; ++r) {
            int row = r + 8 * h;
            myLds[row * 32 + n]      = f32_to_bf16(p0[r]);
            myLds[row * 32 + 16 + n] = f32_to_bf16(p1[r]);
        }
        V16BF ap;                                   // M = n
#pragma unroll
        for (int v = 0; v < 8; ++v) {
            int K = h * 16 + 2 * v;
            ap.u[v] = *(const uint32_t*)(myLds + n * 32 + K);
        }

        // ---- acc += P x V  (4 N-tiles over P) ----
#pragma unroll
        for (int nt = 0; nt < 4; ++nt)
            acc[nt] = __builtin_amdgcn_wmma_f32_16x16x32_bf16(
                false, ap.v, false, bt[nt].v, (short)0, acc[nt], false, false);
    }

    // ---- normalize, round-trip through LDS (16x64), fused x Wo + bo ----
    float invl[8];
#pragma unroll
    for (int r = 0; r < 8; ++r) invl[r] = 1.0f / lrow[r];
#pragma unroll
    for (int nt = 0; nt < 4; ++nt)
#pragma unroll
        for (int r = 0; r < 8; ++r) {
            int row = r + 8 * h;
            myLds[row * 64 + nt * 16 + n] = f32_to_bf16(acc[nt][r] * invl[r]);
        }
    V16BF aw[2];
#pragma unroll
    for (int t = 0; t < 2; ++t)
#pragma unroll
        for (int v = 0; v < 8; ++v) {
            int K = t * 32 + h * 16 + 2 * v;
            aw[t].u[v] = *(const uint32_t*)(myLds + n * 64 + K);
        }

    float* orow_base = out + ((size_t)b * S + sblk * 16) * D;
    for (int dt = 0; dt < 32; ++dt) {              // N = D = 512, 32 tiles
        int d = dt * 16 + n;
        const uint16_t* wrow = WoT + (size_t)d * P;
        V16BF b0, b1;
#pragma unroll
        for (int v = 0; v < 8; ++v) {
            int K = h * 16 + 2 * v;                 // K = p
            b0.u[v] = *(const uint32_t*)(wrow + K);
            b1.u[v] = *(const uint32_t*)(wrow + 32 + K);
        }
        v8f c = {};
        c = __builtin_amdgcn_wmma_f32_16x16x32_bf16(false, aw[0].v, false, b0.v,
                                                    (short)0, c, false, false);
        c = __builtin_amdgcn_wmma_f32_16x16x32_bf16(false, aw[1].v, false, b1.v,
                                                    (short)0, c, false, false);
        float bias = bo[d];
#pragma unroll
        for (int r = 0; r < 8; ++r) {
            int row = r + 8 * h;
            orow_base[(size_t)row * D + d] = c[r] + bias;
        }
    }
}

// ---------------------------------------------------------------------------
extern "C" void kernel_launch(void* const* d_in, const int* in_sizes, int n_in,
                              void* d_out, int out_size, void* d_ws, size_t ws_size,
                              hipStream_t stream) {
    const float* query = (const float*)d_in[0];
    const float* amask = (const float*)d_in[1];
    const float* Wq    = (const float*)d_in[2];
    const float* bq    = (const float*)d_in[3];
    const float* Wk    = (const float*)d_in[4];
    const float* bk    = (const float*)d_in[5];
    const float* Wv    = (const float*)d_in[6];
    const float* bv    = (const float*)d_in[7];
    const float* Wo    = (const float*)d_in[8];
    const float* bo    = (const float*)d_in[9];
    float* out = (float*)d_out;

    char* ws = (char*)d_ws;
    uint16_t* WqT = (uint16_t*)(ws + 0);
    uint16_t* WkT = (uint16_t*)(ws + 65536);
    uint16_t* WvT = (uint16_t*)(ws + 131072);
    uint16_t* WoT = (uint16_t*)(ws + 196608);
    uint16_t* qb  = (uint16_t*)(ws + 262144);                      // 2 MB
    uint16_t* kb  = (uint16_t*)(ws + 262144 + 2097152);            // 2 MB
    uint16_t* vT  = (uint16_t*)(ws + 262144 + 2 * 2097152);        // 2 MB
    uint16_t* qbf = (uint16_t*)(ws + 262144 + 3 * 2097152);        // 16 MB

    prep_query<<<(B * S * D / 4 + 255) / 256, 256, 0, stream>>>(query, qbf);
    prep_weights<<<(D * P + 255) / 256, 256, 0, stream>>>(Wq, Wk, Wv, Wo,
                                                          WqT, WkT, WvT, WoT);
    qkv_kernel<<<384, 256, 0, stream>>>(qbf, WqT, WkT, WvT, bq, bk, bv,
                                        qb, kb, vT);
    attn_kernel<<<256, 128, 0, stream>>>(qb, kb, vT, WoT, bo, amask, out);
}
